// FlowFieldDeformation_1331439862520
// MI455X (gfx1250) — matrix operand order repaired and checked by reading, compiled, not run on hardware
//
#include <hip/hip_runtime.h>
#include <math.h>

// ---------------------------------------------------------------------------
// FlowFieldDeformation for MI455X (gfx1250): bf16 channel-last implicit-GEMM
// convs on v_wmma_f32_16x16x32_bf16 with zero-halo padding (no predication in
// the K-loop), LDS-staged B fragments, VALU for corr/upsample/head/warp.
// ---------------------------------------------------------------------------

typedef unsigned short u16;
typedef __attribute__((ext_vector_type(16))) __bf16 v16bf;
typedef __attribute__((ext_vector_type(8)))  float  v8f;
typedef __attribute__((ext_vector_type(4)))  unsigned int v4u;

union Frag {
  v4u  q[2];
  v16bf v;
  u16  s[16];
};

static constexpr int BB = 8;
static constexpr int HH = 160;
static constexpr int WW = 160;
static constexpr int HW = HH * WW;
static constexpr int HP = (HH + 2) * (WW + 2);   // halo frame pixels (162*162)
static constexpr int Hs = 80;                    // half-res
static constexpr int HWs = Hs * Hs;
static constexpr int NHALO = 2 * (WW + 2) + 2 * HH;  // 644 halo pixels / image

__device__ __forceinline__ u16 f2bf(float f) {
  unsigned u = __float_as_uint(f);
  unsigned r = (u + 0x7FFFu + ((u >> 16) & 1u)) >> 16;
  return (u16)r;
}
__device__ __forceinline__ float bf2f(u16 s) {
  return __uint_as_float(((unsigned)s) << 16);
}
__device__ __forceinline__ float lrelu(float v) { return v > 0.f ? v : 0.1f * v; }

// ---------------------------------------------------------------------------
// 0) Zero the 1-pixel halo ring of a channel-last halo buffer.
// ---------------------------------------------------------------------------
__global__ void zero_halo_kernel(u16* __restrict__ buf, int C) {
  int idx = blockIdx.x * 256 + threadIdx.x;
  int total = BB * NHALO * C;
  if (idx >= total) return;
  int c = idx % C;
  int r = idx / C;
  int pos = r % NHALO;
  int b = r / NHALO;
  int y, x;
  if (pos < WW + 2) {             // top row y=-1, x=-1..160
    y = -1; x = pos - 1;
  } else if (pos < 2 * (WW + 2)) {  // bottom row y=160
    y = HH; x = pos - (WW + 2) - 1;
  } else {                        // left/right columns, y=0..159
    int p = pos - 2 * (WW + 2);
    y = p >> 1;
    x = (p & 1) ? WW : -1;
  }
  buf[((size_t)(b * (HH + 2) + y + 1) * (WW + 2) + (x + 1)) * C + c] = 0;
}

// ---------------------------------------------------------------------------
// 1) Transposed conv 4x4 stride2 pad1, per-channel (groups==in_c, opg==1).
//    out[o] = sum_t in[(o+1-t)/2] * w[t], (o+1-t) even & in-range.
// ---------------------------------------------------------------------------
__global__ void upsample_kernel(const float* __restrict__ flow_in,
                                const float* __restrict__ conf_in,
                                const float* __restrict__ up_conf_w,
                                const float* __restrict__ up_flow_w,
                                float* __restrict__ flow_up,
                                float* __restrict__ conf_up) {
  int idx = blockIdx.x * 256 + threadIdx.x;
  if (idx >= BB * HW) return;
  int x = idx % WW;
  int y = (idx / WW) % HH;
  int b = idx / HW;
  float ca = 0.f, f0 = 0.f, f1 = 0.f;
#pragma unroll
  for (int ky = 0; ky < 4; ++ky) {
    int ty = y + 1 - ky;
    if (ty < 0 || (ty & 1)) continue;
    int iy = ty >> 1;
    if (iy >= Hs) continue;
#pragma unroll
    for (int kx = 0; kx < 4; ++kx) {
      int tx = x + 1 - kx;
      if (tx < 0 || (tx & 1)) continue;
      int ix = tx >> 1;
      if (ix >= Hs) continue;
      int ii = iy * Hs + ix;
      int kk = ky * 4 + kx;
      ca = fmaf(conf_in[b * HWs + ii],            up_conf_w[kk],      ca);
      f0 = fmaf(flow_in[(b * 2 + 0) * HWs + ii],  up_flow_w[kk],      f0);
      f1 = fmaf(flow_in[(b * 2 + 1) * HWs + ii],  up_flow_w[16 + kk], f1);
    }
  }
  conf_up[idx] = ca;
  flow_up[(b * 2 + 0) * HW + y * WW + x] = f0;
  flow_up[(b * 2 + 1) * HW + y * WW + x] = f1;
}

// ---------------------------------------------------------------------------
// 2) Self-correlation (7x7 taps, dilation 2, C=128) -> lrelu -> /128, fused
//    with conf concat; writes x0 as bf16 channel-last HALO [B][162][162][64].
//    One 160-thread block per (b,y); lane == pixel x (coalesced shifted loads).
// ---------------------------------------------------------------------------
__global__ __launch_bounds__(160)
void corr_kernel(const float* __restrict__ feats,
                 const float* __restrict__ conf_up,
                 u16* __restrict__ x0) {
  int x = threadIdx.x;
  int y = blockIdx.x % HH;
  int b = blockIdx.x / HH;
  const float* f = feats + (size_t)(b * 2) * 128 * HW;  // feats[:,0]
  float acc[49];
#pragma unroll
  for (int o = 0; o < 49; ++o) acc[o] = 0.f;
  for (int c = 0; c < 128; ++c) {
    const float* fc = f + (size_t)c * HW;
    float v0 = fc[y * WW + x];
#pragma unroll
    for (int o = 0; o < 49; ++o) {
      int dy = (o / 7 - 3) * 2;
      int dx = (o % 7 - 3) * 2;
      int ys = y + dy;
      int xs = x + dx;
      float s = (ys >= 0 && ys < HH && xs >= 0 && xs < WW) ? fc[ys * WW + xs] : 0.f;
      acc[o] = fmaf(v0, s, acc[o]);
    }
  }
  u16* dst = x0 + ((size_t)(b * (HH + 2) + y + 1) * (WW + 2) + (x + 1)) * 64;
#pragma unroll
  for (int o = 0; o < 49; ++o) dst[o] = f2bf(lrelu(acc[o]) * (1.f / 128.f));
  dst[49] = f2bf(conf_up[(size_t)b * HW + y * WW + x]);
#pragma unroll
  for (int c = 50; c < 64; ++c) dst[c] = 0;
}

// ---------------------------------------------------------------------------
// 3) Pack conv weights (fp32 [Cout][Cin][3][3]) into bf16 B-fragment layout:
//    wf[octile][tap][kslice][lane][e], lane n = lane&15, K = 8*(lane>>4)+e (e<8), 16+8*(lane>>4)+(e-8) (e>=8).
// ---------------------------------------------------------------------------
__global__ void pack_w_kernel(const float* __restrict__ w, u16* __restrict__ wf,
                              int Cout, int Cin, int CinPad) {
  int ksn = CinPad / 32;
  int total = (Cout / 16) * 9 * ksn * 512;
  int idx = blockIdx.x * 256 + threadIdx.x;
  if (idx >= total) return;
  int e = idx & 15;
  int lane = (idx >> 4) & 31;
  int r = idx >> 9;
  int ks = r % ksn;  r /= ksn;
  int tap = r % 9;
  int octile = r / 9;
  int oc = octile * 16 + (lane & 15);
  int c = ks * 32 + (e < 8 ? (lane >> 4) * 8 + e : 16 + (lane >> 4) * 8 + (e - 8));
  int ky = tap / 3, kx = tap % 3;
  float v = (c < Cin) ? w[((size_t)(oc * Cin + c) * 3 + ky) * 3 + kx] : 0.f;
  wf[idx] = f2bf(v);
}

// ---------------------------------------------------------------------------
// 4) 3x3 conv as implicit GEMM with v_wmma_f32_16x16x32_bf16.
//    Halo input => NO bounds checks, NO exec manipulation in the K-loop:
//    pure ds_load_b128 + global_load_b128 + wmma. Block = 160 threads =
//    5 waves covering a full row; each wave: two 16-px tiles x 16 oc.
// ---------------------------------------------------------------------------
template <int CINP, int COUT, bool OUT_HALO>
__global__ __launch_bounds__(160)
void conv3x3_wmma_kernel(const u16* __restrict__ in,
                         const u16* __restrict__ wf,
                         const float* __restrict__ bias,
                         u16* __restrict__ out) {
  constexpr int KS = CINP / 32;
  constexpr int OCT = COUT / 16;
  __shared__ u16 smem[9 * KS * 512];

  int octile = blockIdx.x % OCT;
  int y = (blockIdx.x / OCT) % HH;
  int b = blockIdx.x / (OCT * HH);

  const u16* wsrc = wf + (size_t)octile * 9 * KS * 512;
  for (int i = threadIdx.x; i < 9 * KS * 512; i += 160) smem[i] = wsrc[i];
  __syncthreads();

  int lane = threadIdx.x & 31;
  int wave = threadIdx.x >> 5;   // 0..4, covers x in [32*wave, 32*wave+32)
  int m = lane & 15;
  int half = lane >> 4;
  int xA = wave * 32 + m;        // first 16-px tile
  int xB = xA + 16;              // second 16-px tile

  v8f acc0 = {0.f, 0.f, 0.f, 0.f, 0.f, 0.f, 0.f, 0.f};
  v8f acc1 = {0.f, 0.f, 0.f, 0.f, 0.f, 0.f, 0.f, 0.f};

#pragma unroll
  for (int tap = 0; tap < 9; ++tap) {
    int ky = tap / 3, kx = tap % 3;
    int ys = y + ky - 1;
    // halo coords: row index ys+1, col index xs+1 — always in-bounds
    const u16* rowp =
        in + (size_t)(b * (HH + 2) + ys + 1) * (WW + 2) * CINP;
    const u16* apA = rowp + (size_t)(xA + kx) * CINP + half * 8;
    const u16* apB = rowp + (size_t)(xB + kx) * CINP + half * 8;
#pragma unroll
    for (int ks = 0; ks < KS; ++ks) {
      Frag Bf;
      const u16* bp = smem + ((tap * KS + ks) * 32 + lane) * 16;
      Bf.q[0] = *(const v4u*)(bp);
      Bf.q[1] = *(const v4u*)(bp + 8);

      Frag Aa;
      Aa.q[0] = *(const v4u*)(apA + ks * 32);
      Aa.q[1] = *(const v4u*)(apA + ks * 32 + 16);
      acc0 = __builtin_amdgcn_wmma_f32_16x16x32_bf16(
          false, Aa.v, false, Bf.v, (short)0, acc0, false, false);

      Frag Ab;
      Ab.q[0] = *(const v4u*)(apB + ks * 32);
      Ab.q[1] = *(const v4u*)(apB + ks * 32 + 16);
      acc1 = __builtin_amdgcn_wmma_f32_16x16x32_bf16(
          false, Ab.v, false, Bf.v, (short)0, acc1, false, false);
    }
  }

  float bia = bias[octile * 16 + m];   // n == lane&15 == m
  size_t obase;
  if (OUT_HALO) {
    obase = ((size_t)(b * (HH + 2) + y + 1) * (WW + 2) + 1) * COUT +
            octile * 16 + m;
  } else {
    obase = (size_t)(b * HH + y) * WW * COUT + octile * 16 + m;
  }
#pragma unroll
  for (int v = 0; v < 8; ++v) {
    int mr = v + half * 8;
    float r0 = lrelu(acc0[v] + bia);
    out[obase + (size_t)(wave * 32 + mr) * COUT] = f2bf(r0);
    float r1 = lrelu(acc1[v] + bia);
    out[obase + (size_t)(wave * 32 + 16 + mr) * COUT] = f2bf(r1);
  }
}

// ---------------------------------------------------------------------------
// 5) 5x5 heads (disp 2ch + conf 1ch) over x3 (bf16, channel-last 32, no halo),
//    fused with bilinear warp of flow_up and sigmoid on conf.
// ---------------------------------------------------------------------------
__global__ void head_warp_kernel(const u16* __restrict__ x3,
                                 const float* __restrict__ flow_up,
                                 const float* __restrict__ dw,
                                 const float* __restrict__ db,
                                 const float* __restrict__ cw,
                                 const float* __restrict__ cb,
                                 float* __restrict__ outp) {
  int idx = blockIdx.x * 256 + threadIdx.x;
  if (idx >= BB * HW) return;
  int x = idx % WW;
  int y = (idx / WW) % HH;
  int b = idx / HW;

  float a0 = 0.f, a1 = 0.f, ac = 0.f;
  for (int t = 0; t < 25; ++t) {
    int ys = y + t / 5 - 2;
    int xs = x + t % 5 - 2;
    if (ys < 0 || ys >= HH || xs < 0 || xs >= WW) continue;
    const u16* p = x3 + ((size_t)(b * HH + ys) * WW + xs) * 32;
#pragma unroll
    for (int c = 0; c < 32; ++c) {
      float v = bf2f(p[c]);
      a0 = fmaf(v, dw[(0 * 32 + c) * 25 + t], a0);
      a1 = fmaf(v, dw[(1 * 32 + c) * 25 + t], a1);
      ac = fmaf(v, cw[c * 25 + t], ac);
    }
  }
  a0 += db[0];
  a1 += db[1];
  ac += cb[0];

  // bilinear warp of flow_up by disp (a0, a1) with reference masking
  float px = (float)x + a0;
  float py = (float)y + a1;
  float x0f = floorf(px), y0f = floorf(py);
  float wx1 = px - x0f, wx0 = 1.f - wx1;
  float wy1 = py - y0f, wy0 = 1.f - wy1;
  float cwt[4] = {wy0 * wx0, wy0 * wx1, wy1 * wx0, wy1 * wx1};
  float cx[4] = {x0f, x0f + 1.f, x0f, x0f + 1.f};
  float cy[4] = {y0f, y0f, y0f + 1.f, y0f + 1.f};

  float v0 = 0.f, v1 = 0.f, msk = 0.f;
  const float* fu0 = flow_up + (size_t)(b * 2 + 0) * HW;
  const float* fu1 = flow_up + (size_t)(b * 2 + 1) * HW;
#pragma unroll
  for (int k = 0; k < 4; ++k) {
    float xi = cx[k], yi = cy[k];
    bool valid = (xi >= 0.f) && (xi <= (float)(WW - 1)) &&
                 (yi >= 0.f) && (yi <= (float)(HH - 1));
    int xc = (int)fminf(fmaxf(xi, 0.f), (float)(WW - 1));
    int yc = (int)fminf(fmaxf(yi, 0.f), (float)(HH - 1));
    int ii = yc * WW + xc;
    if (valid) {
      v0 = fmaf(cwt[k], fu0[ii], v0);
      v1 = fmaf(cwt[k], fu1[ii], v1);
      msk += cwt[k];
    }
  }
  float mask = (msk >= 1.f) ? 1.f : 0.f;

  outp[(size_t)(b * 2 + 0) * HW + y * WW + x] = v0 * mask;
  outp[(size_t)(b * 2 + 1) * HW + y * WW + x] = v1 * mask;
  outp[(size_t)BB * 2 * HW + (size_t)b * HW + y * WW + x] =
      1.f / (1.f + expf(-ac));
}

// ---------------------------------------------------------------------------
// Host launcher
// ---------------------------------------------------------------------------
extern "C" void kernel_launch(void* const* d_in, const int* in_sizes, int n_in,
                              void* d_out, int out_size, void* d_ws,
                              size_t ws_size, hipStream_t stream) {
  (void)in_sizes; (void)n_in; (void)out_size; (void)ws_size;

  const float* feats     = (const float*)d_in[0];
  const float* flow      = (const float*)d_in[1];
  const float* conf      = (const float*)d_in[2];
  const float* up_conf_w = (const float*)d_in[3];
  const float* up_flow_w = (const float*)d_in[4];
  const float* w1 = (const float*)d_in[5];
  const float* b1 = (const float*)d_in[6];
  const float* w2 = (const float*)d_in[7];
  const float* b2 = (const float*)d_in[8];
  const float* w3 = (const float*)d_in[9];
  const float* b3 = (const float*)d_in[10];
  const float* dw = (const float*)d_in[11];
  const float* db = (const float*)d_in[12];
  const float* cw = (const float*)d_in[13];
  const float* cb = (const float*)d_in[14];
  float* out = (float*)d_out;

  // workspace carving (256B aligned)
  char* ws = (char*)d_ws;
  size_t off = 0;
  auto carve = [&](size_t bytes) -> char* {
    char* p = ws + off;
    off += (bytes + 255) & ~(size_t)255;
    return p;
  };
  float* flow_up = (float*)carve((size_t)BB * 2 * HW * 4);
  float* conf_up = (float*)carve((size_t)BB * HW * 4);
  u16* x0 = (u16*)carve((size_t)BB * HP * 64 * 2);    // halo
  u16* x1 = (u16*)carve((size_t)BB * HP * 128 * 2);   // halo
  u16* x2 = (u16*)carve((size_t)BB * HP * 64 * 2);    // halo
  u16* x3 = (u16*)carve((size_t)BB * HW * 32 * 2);    // no halo
  u16* wf1 = (u16*)carve((size_t)8 * 9 * 2 * 512 * 2);
  u16* wf2 = (u16*)carve((size_t)4 * 9 * 4 * 512 * 2);
  u16* wf3 = (u16*)carve((size_t)2 * 9 * 2 * 512 * 2);

  const int npix = BB * HW;  // 204800

  // zero the halo rings (interior is fully overwritten each call)
  zero_halo_kernel<<<dim3((BB * NHALO * 64 + 255) / 256), dim3(256), 0,
                     stream>>>(x0, 64);
  zero_halo_kernel<<<dim3((BB * NHALO * 128 + 255) / 256), dim3(256), 0,
                     stream>>>(x1, 128);
  zero_halo_kernel<<<dim3((BB * NHALO * 64 + 255) / 256), dim3(256), 0,
                     stream>>>(x2, 64);

  upsample_kernel<<<dim3((npix + 255) / 256), dim3(256), 0, stream>>>(
      flow, conf, up_conf_w, up_flow_w, flow_up, conf_up);

  pack_w_kernel<<<dim3((8 * 9 * 2 * 512 + 255) / 256), dim3(256), 0, stream>>>(
      w1, wf1, 128, 50, 64);
  pack_w_kernel<<<dim3((4 * 9 * 4 * 512 + 255) / 256), dim3(256), 0, stream>>>(
      w2, wf2, 64, 128, 128);
  pack_w_kernel<<<dim3((2 * 9 * 2 * 512 + 255) / 256), dim3(256), 0, stream>>>(
      w3, wf3, 32, 64, 64);

  corr_kernel<<<dim3(BB * HH), dim3(160), 0, stream>>>(feats, conf_up, x0);

  conv3x3_wmma_kernel<64, 128, true>
      <<<dim3(BB * HH * 8), dim3(160), 0, stream>>>(x0, wf1, b1, x1);
  conv3x3_wmma_kernel<128, 64, true>
      <<<dim3(BB * HH * 4), dim3(160), 0, stream>>>(x1, wf2, b2, x2);
  conv3x3_wmma_kernel<64, 32, false>
      <<<dim3(BB * HH * 2), dim3(160), 0, stream>>>(x2, wf3, b3, x3);

  head_warp_kernel<<<dim3((npix + 255) / 256), dim3(256), 0, stream>>>(
      x3, flow_up, dw, db, cw, cb, out);
}
